// GRU_54606214202224
// MI455X (gfx1250) — matrix operand order repaired and checked by reading, not compiled, not run on hardware
//
#include <hip/hip_runtime.h>
#include <hip/hip_bf16.h>

// ---------------------------------------------------------------------------
// CDNA5 (gfx1250) bf16-WMMA GRU. All matmuls use v_wmma_f32_16x16x32_bf16.
// Weights converted once to bf16 + transposed [N,K]. Recurrent activations
// stay f32 (converted on the fly in the A-fragment builder); the top-layer
// output is additionally emitted in bf16 so the big logits GEMM reads bf16
// A directly (no per-pass conversion, half the A traffic).
// ---------------------------------------------------------------------------

typedef __attribute__((ext_vector_type(16))) __bf16         v16bf;
typedef __attribute__((ext_vector_type(16))) unsigned short v16us;
typedef __attribute__((ext_vector_type(8)))  unsigned short v8us;
typedef __attribute__((ext_vector_type(8)))  float          v8f;

#define SS 256
#define BB 64
#define VV 10000
#define EE 512
#define HH 1024
#define NPAD 10112   // V rounded up to multiple of 128

__device__ __forceinline__ unsigned short f2bf(float f) {
    union { float f; unsigned u; } x; x.f = f;
    unsigned r = x.u + 0x7FFFu + ((x.u >> 16) & 1u);   // round-to-nearest-even
    return (unsigned short)(r >> 16);
}

// --- A fragment: 16x32 bf16. Per-lane: row = lane&15, K runs
//     [kb..kb+7] and [kb+16..kb+23] with kb = kbase + (lane>>4)*8.
// AMODE 0: A = X[M,K] f32
// AMODE 1: A = concat(X[M,K1] f32, H f32)     (K1 % 32 == 0 -> runs don't straddle)
// AMODE 2: A = concat(X[M,K1] f32, R*H f32)
// AMODE 3: A = X[M,K] bf16 (direct, no conversion)
template<int AMODE>
__device__ __forceinline__ v16bf load_a_frag(
    const void* __restrict__ Xv, int ldx, int K1,
    const float* __restrict__ Hs, int ldh,
    const float* __restrict__ Rs, int ldr,
    int row, int kbase, int lane)
{
    int kb = kbase + ((lane >> 4) << 3);
    if constexpr (AMODE == 3) {
        const unsigned short* Xb = (const unsigned short*)Xv;
        union { v8us h[2]; v16bf bf; } u;
        u.h[0] = *(const v8us*)(Xb + (size_t)row * ldx + kb);
        u.h[1] = *(const v8us*)(Xb + (size_t)row * ldx + kb + 16);
        return u.bf;
    } else {
        const float* X = (const float*)Xv;
        union { v16us us; v16bf bf; } cvt;
#pragma unroll
        for (int run = 0; run < 2; ++run) {
            int k0 = kb + run * 16;
            float vals[8];
            bool fromX = (AMODE == 0) || (k0 < K1);
            if (fromX) {
                const float4* p = (const float4*)(X + (size_t)row * ldx + k0);
                float4 a = p[0], b = p[1];
                vals[0]=a.x; vals[1]=a.y; vals[2]=a.z; vals[3]=a.w;
                vals[4]=b.x; vals[5]=b.y; vals[6]=b.z; vals[7]=b.w;
            } else {
                int kk = k0 - K1;
                const float4* ph = (const float4*)(Hs + (size_t)row * ldh + kk);
                float4 a = ph[0], b = ph[1];
                vals[0]=a.x; vals[1]=a.y; vals[2]=a.z; vals[3]=a.w;
                vals[4]=b.x; vals[5]=b.y; vals[6]=b.z; vals[7]=b.w;
                if (AMODE == 2) {
                    const float4* pr = (const float4*)(Rs + (size_t)row * ldr + kk);
                    float4 c = pr[0], d = pr[1];
                    vals[0]*=c.x; vals[1]*=c.y; vals[2]*=c.z; vals[3]*=c.w;
                    vals[4]*=d.x; vals[5]*=d.y; vals[6]*=d.z; vals[7]*=d.w;
                }
            }
#pragma unroll
            for (int j = 0; j < 8; ++j) cvt.us[run * 8 + j] = f2bf(vals[j]);
        }
        return cvt.bf;
    }
}

// --- B fragment: 32x16 bf16 from pre-transposed Wt[N,K]; col = lane&15,
//     K runs [kb..kb+7] and [kb+16..kb+23], kb = kbase + (lane>>4)*8 -> two aligned 16B loads.
__device__ __forceinline__ v16bf load_b_frag(
    const unsigned short* __restrict__ Bt, int K, int nbase, int kbase, int lane)
{
    int col = nbase + (lane & 15);
    int kb  = kbase + ((lane >> 4) << 3);
    union { v8us h[2]; v16bf bf; } cvt;
    cvt.h[0] = *(const v8us*)(Bt + (size_t)col * K + kb);
    cvt.h[1] = *(const v8us*)(Bt + (size_t)col * K + kb + 16);
    return cvt.bf;
}

// ---------------------------------------------------------------------------
// Generic WMMA GEMM: C[M,N] = act(A[M,K] x B[K,N] + bias), bf16 in / f32 acc.
// 256 threads = 8 waves arranged MW m-waves x (8/MW) n-groups; each wave owns
// NT 16x16 accumulator tiles.  BM = MW*16, BN = (8/MW)*NT*16.
// EPI 0: sigmoid -> out0                                  (r|z fused GEMM)
// EPI 1: h = (1-z)*hprev + z*tanh(acc+b) -> out0 f32 (+out1 bf16)  (GRU update)
// EPI 2: acc + bias -> out0, guarded by n < Nreal         (logits)
// ---------------------------------------------------------------------------
template<int AMODE, int EPI, int MW, int NT>
__global__ __launch_bounds__(256) void gemm_wmma(
    const void* __restrict__ Xv, int ldx, int K1,
    const float* __restrict__ Hs, int ldh,
    const float* __restrict__ Rs, int ldr,
    const unsigned short* __restrict__ Bt, int K,
    const float* __restrict__ bias,
    float* __restrict__ out0, int ldo0,
    void* __restrict__ out1v, int ldo1,
    const float* __restrict__ Zs, int ldz,
    const float* __restrict__ Hprev, int ldhp,
    int Nreal)
{
    int lane  = threadIdx.x & 31;
    int wid   = threadIdx.x >> 5;
    int mwave = wid % MW;
    int nwave = wid / MW;
    int mbase = blockIdx.y * (MW * 16) + mwave * 16;
    int nbase = blockIdx.x * ((8 / MW) * NT * 16) + nwave * (NT * 16);
    int row   = mbase + (lane & 15);

    v8f acc[NT] = {};

    for (int kb = 0; kb < K; kb += 32) {
        v16bf a = load_a_frag<AMODE>(Xv, ldx, K1, Hs, ldh, Rs, ldr, row, kb, lane);
        v16bf bfr[NT];
#pragma unroll
        for (int j = 0; j < NT; ++j)
            bfr[j] = load_b_frag(Bt, K, nbase + j * 16, kb, lane);
#pragma unroll
        for (int j = 0; j < NT; ++j)
            acc[j] = __builtin_amdgcn_wmma_f32_16x16x32_bf16(
                false, a, false, bfr[j], (short)0, acc[j], false, false);
    }

    // C tile layout: VGPR e -> M = mbase + (lane>>4)*8 + e, N = nbase + (lane&15)
    int mrow = mbase + ((lane >> 4) << 3);
#pragma unroll
    for (int j = 0; j < NT; ++j) {
        int n = nbase + j * 16 + (lane & 15);
        float bv = (EPI == 2) ? (n < Nreal ? bias[n] : 0.0f) : bias[n];
#pragma unroll
        for (int e = 0; e < 8; ++e) {
            int m = mrow + e;
            float v = acc[j][e] + bv;
            if (EPI == 0) {
                out0[(size_t)m * ldo0 + n] = 1.0f / (1.0f + __expf(-v));
            } else if (EPI == 1) {
                float hh = tanhf(v);
                float z  = Zs[(size_t)m * ldz + n];
                float hp = Hprev[(size_t)m * ldhp + n];
                float hn = (1.0f - z) * hp + z * hh;
                out0[(size_t)m * ldo0 + n] = hn;
                unsigned short* out1 = (unsigned short*)out1v;
                if (out1) out1[(size_t)m * ldo1 + n] = f2bf(hn);
            } else {
                if (n < Nreal) out0[(size_t)m * ldo0 + n] = v;
            }
        }
    }
}

// ------------------------- setup / utility kernels -------------------------

__global__ void gather_emb(const int* __restrict__ inputs,
                           const float* __restrict__ Emb,
                           float* __restrict__ out)
{
    int tok = blockIdx.x;                 // S*B tokens
    int idx = inputs[tok];
    const float* src = Emb + (size_t)idx * EE;
    float* dst = out + (size_t)tok * EE;
    for (int i = threadIdx.x; i < EE; i += blockDim.x) dst[i] = src[i];
}

// pack [K,Nsrc] f32 row-major -> bf16 transposed [Npad,K], zero-pad n>=Nsrc
__global__ void pack_w(const float* __restrict__ src,
                       unsigned short* __restrict__ Wt,
                       int K, int Nsrc, int Npad)
{
    size_t i = (size_t)blockIdx.x * blockDim.x + threadIdx.x;
    size_t total = (size_t)Npad * K;
    if (i >= total) return;
    int n = (int)(i / K), k = (int)(i % K);
    float v = (n < Nsrc) ? src[(size_t)k * Nsrc + n] : 0.0f;
    Wt[i] = f2bf(v);
}

// pack Wr|Wz -> bf16 transposed [2H,K]; also builds concat bias brz[2H]
__global__ void pack_rz(const float* __restrict__ Wr, const float* __restrict__ Wz,
                        const float* __restrict__ br, const float* __restrict__ bz,
                        unsigned short* __restrict__ Wt, float* __restrict__ brz,
                        int K)
{
    size_t i = (size_t)blockIdx.x * blockDim.x + threadIdx.x;
    size_t total = (size_t)2 * HH * K;
    if (i >= total) return;
    int n = (int)(i / K), k = (int)(i % K);
    float v = (n < HH) ? Wr[(size_t)k * HH + n] : Wz[(size_t)k * HH + (n - HH)];
    Wt[i] = f2bf(v);
    if (i < (size_t)2 * HH) brz[i] = (i < HH) ? br[i] : bz[i - HH];
}

__global__ void copyf(const float* __restrict__ src, float* __restrict__ dst, int n)
{
    int i = blockIdx.x * blockDim.x + threadIdx.x;
    if (i < n) dst[i] = src[i];
}

// ---------------------------------------------------------------------------

extern "C" void kernel_launch(void* const* d_in, const int* in_sizes, int n_in,
                              void* d_out, int out_size, void* d_ws, size_t ws_size,
                              hipStream_t stream)
{
    (void)in_sizes; (void)n_in; (void)out_size; (void)ws_size;

    const int*   inputs = (const int*)  d_in[0];
    const float* hidden = (const float*)d_in[1];
    const float* Emb    = (const float*)d_in[2];
    const float* Wr0 = (const float*)d_in[3];  const float* br0 = (const float*)d_in[4];
    const float* Wz0 = (const float*)d_in[5];  const float* bz0 = (const float*)d_in[6];
    const float* Wh0 = (const float*)d_in[7];  const float* bh0 = (const float*)d_in[8];
    const float* Wr1 = (const float*)d_in[9];  const float* br1 = (const float*)d_in[10];
    const float* Wz1 = (const float*)d_in[11]; const float* bz1 = (const float*)d_in[12];
    const float* Wh1 = (const float*)d_in[13]; const float* bh1 = (const float*)d_in[14];
    const float* Wout = (const float*)d_in[15]; const float* bout = (const float*)d_in[16];

    const int K0 = EE + HH;   // 1536
    const int K1 = 2 * HH;    // 2048
    const int BH = BB * HH;

    // ---- workspace carve (256B aligned slices) ----
    char* p = (char*)d_ws;
    auto carve = [&](size_t bytes) -> void* {
        void* r = (void*)p;
        p += (bytes + 255) & ~(size_t)255;
        return r;
    };
    float*          emb_f32  = (float*)carve((size_t)SS * BB * EE * 4);
    unsigned short* top_bf16 = (unsigned short*)carve((size_t)SS * BB * HH * 2);
    unsigned short* Wrz0t    = (unsigned short*)carve((size_t)2 * HH * K0 * 2);
    unsigned short* Wh0t     = (unsigned short*)carve((size_t)HH * K0 * 2);
    unsigned short* Wrz1t    = (unsigned short*)carve((size_t)2 * HH * K1 * 2);
    unsigned short* Wh1t     = (unsigned short*)carve((size_t)HH * K1 * 2);
    unsigned short* Woutt    = (unsigned short*)carve((size_t)NPAD * HH * 2);
    float*          brz0     = (float*)carve((size_t)2 * HH * 4);
    float*          brz1     = (float*)carve((size_t)2 * HH * 4);
    float*          h0buf    = (float*)carve((size_t)2 * BH * 4);   // ping-pong
    float*          h1buf    = (float*)carve((size_t)2 * BH * 4);
    float*          rz0      = (float*)carve((size_t)BB * 2 * HH * 4);
    float*          rz1      = (float*)carve((size_t)BB * 2 * HH * 4);

    // ---- setup: embedding gather, weight convert/transpose, state init ----
    gather_emb<<<SS * BB, 256, 0, stream>>>(inputs, Emb, emb_f32);

    auto blks = [](size_t total) { return (unsigned)((total + 255) / 256); };
    pack_rz<<<blks((size_t)2 * HH * K0), 256, 0, stream>>>(Wr0, Wz0, br0, bz0, Wrz0t, brz0, K0);
    pack_w <<<blks((size_t)HH * K0),     256, 0, stream>>>(Wh0, Wh0t, K0, HH, HH);
    pack_rz<<<blks((size_t)2 * HH * K1), 256, 0, stream>>>(Wr1, Wz1, br1, bz1, Wrz1t, brz1, K1);
    pack_w <<<blks((size_t)HH * K1),     256, 0, stream>>>(Wh1, Wh1t, K1, HH, HH);
    pack_w <<<blks((size_t)NPAD * HH),   256, 0, stream>>>(Wout, Woutt, HH, VV, NPAD);

    copyf<<<blks(BH), 256, 0, stream>>>(hidden,      h0buf, BH);
    copyf<<<blks(BH), 256, 0, stream>>>(hidden + BH, h1buf, BH);

    // ---- recurrent scan: 4 fused WMMA GEMMs per timestep (BM=64, BN=128) ----
    for (int t = 0; t < SS; ++t) {
        int cur = t & 1, nxt = cur ^ 1;
        const float* xt = emb_f32 + (size_t)t * BB * EE;
        float* h0c = h0buf + (size_t)cur * BH;  float* h0n = h0buf + (size_t)nxt * BH;
        float* h1c = h1buf + (size_t)cur * BH;  float* h1n = h1buf + (size_t)nxt * BH;

        // L0: rz = sigmoid([x,h0] @ Wrz + brz)
        gemm_wmma<1, 0, 4, 4><<<dim3(2 * HH / 128, 1), 256, 0, stream>>>(
            xt, EE, EE, h0c, HH, nullptr, 0,
            Wrz0t, K0, brz0, rz0, 2 * HH,
            nullptr, 0, nullptr, 0, nullptr, 0, 2 * HH);
        // L0: h0n = (1-z)h0 + z*tanh([x, r*h0] @ Wh + bh)
        gemm_wmma<2, 1, 4, 4><<<dim3(HH / 128, 1), 256, 0, stream>>>(
            xt, EE, EE, h0c, HH, rz0, 2 * HH,
            Wh0t, K0, bh0, h0n, HH,
            nullptr, 0, rz0 + HH, 2 * HH, h0c, HH, HH);

        // L1: rz = sigmoid([h0n,h1] @ Wrz + brz)
        gemm_wmma<1, 0, 4, 4><<<dim3(2 * HH / 128, 1), 256, 0, stream>>>(
            h0n, HH, HH, h1c, HH, nullptr, 0,
            Wrz1t, K1, brz1, rz1, 2 * HH,
            nullptr, 0, nullptr, 0, nullptr, 0, 2 * HH);
        // L1: h1n = GRU update (f32 state) + bf16 top[t] for logits GEMM
        gemm_wmma<2, 1, 4, 4><<<dim3(HH / 128, 1), 256, 0, stream>>>(
            h0n, HH, HH, h1c, HH, rz1, 2 * HH,
            Wh1t, K1, bh1, h1n, HH,
            top_bf16 + (size_t)t * BH, HH, rz1 + HH, 2 * HH, h1c, HH, HH);
    }

    // final states: t=255 wrote parity 0
    float* logits = (float*)d_out;
    float* hfin   = logits + (size_t)SS * BB * VV;
    copyf<<<blks(BH), 256, 0, stream>>>(h0buf, hfin,      BH);
    copyf<<<blks(BH), 256, 0, stream>>>(h1buf, hfin + BH, BH);

    // logits = top @ Wout + bout   (M=16384, N=10000 pad 10112, K=1024)
    // BM=128 (8 m-waves), BN=128 (8 n-subtiles/wave), bf16 A direct.
    gemm_wmma<3, 2, 8, 8><<<dim3(NPAD / 128, SS * BB / 128), 256, 0, stream>>>(
        top_bf16, HH, HH, nullptr, 0, nullptr, 0,
        Woutt, HH, bout, logits, VV,
        nullptr, 0, nullptr, 0, nullptr, 0, VV);
}
